// MultiHeadAttention_455266533859
// MI455X (gfx1250) — compile-verified, matrix operand order checked by reading, not run on hardware
//
#include <hip/hip_runtime.h>
#include <hip/hip_bf16.h>
#include <stdint.h>

// ---------------------------------------------------------------------------
// Types for CDNA5 WMMA (wave32): v16bf A/B fragments, v8f accumulator.
// ---------------------------------------------------------------------------
typedef __attribute__((ext_vector_type(16))) __bf16 v16bf;
typedef __attribute__((ext_vector_type(8)))  float  v8f;

#define N_NODES 8192
#define H_CH    256
#define HEADS   4
#define HEAD_DIM 64
#define N_EDGES 262144
#define MAXD    512          // per-row neighbor cap (Poisson(32): max deg << 512)

// ---------------------------------------------------------------------------
// helpers
// ---------------------------------------------------------------------------
__device__ __forceinline__ float bf2f(uint32_t h) {
    union { uint32_t u; float f; } c; c.u = h << 16; return c.f;
}
__device__ __forceinline__ uint16_t f2bf(float f) {
    union { float f; uint32_t u; } c; c.f = f;
    uint32_t u = c.u;
    uint32_t r = (u + 0x7FFFu + ((u >> 16) & 1u)) >> 16;   // round-to-nearest-even
    return (uint16_t)r;
}

// ---------------------------------------------------------------------------
// elementwise f32 -> bf16
// ---------------------------------------------------------------------------
__global__ void cvt_f32_bf16(const float* __restrict__ in, uint16_t* __restrict__ out, int n) {
    int i = blockIdx.x * blockDim.x + threadIdx.x;
    if (i < n) out[i] = f2bf(in[i]);
}

// ---------------------------------------------------------------------------
// zero the adjacency bitmask (must be re-zeroed every call: harness poisons ws)
// ---------------------------------------------------------------------------
__global__ void zero_u32(uint32_t* __restrict__ p, int n) {
    for (int i = blockIdx.x * blockDim.x + threadIdx.x; i < n; i += gridDim.x * blockDim.x)
        p[i] = 0u;
}

// ---------------------------------------------------------------------------
// scatter edges into the bitmask adjacency: adj[row][col] = 1  (dedupes)
// edge_index is int64 [2, E]
// ---------------------------------------------------------------------------
__global__ void fill_adj(const long long* __restrict__ eidx, uint32_t* __restrict__ adj) {
    int e = blockIdx.x * blockDim.x + threadIdx.x;
    if (e >= N_EDGES) return;
    int row = (int)eidx[e];
    int col = (int)eidx[N_EDGES + e];
    atomicOr(&adj[row * (H_CH) + (col >> 5)], 1u << (col & 31));
}

// ---------------------------------------------------------------------------
// WMMA GEMM:  out[M,256] = A_bf16[M,256] @ W_bf16[256,256]^T + bias
//   One 16x16 output tile per wave, K-loop 8 x (16x16x32 bf16 WMMA).
//   outMode 0 -> fp32 output, 1 -> bf16 output.
// Fragment layouts per CDNA5 ISA 7.12.2 (wave32):
//   A: lane L holds row (L&15); halves 0..7 = K {base..base+7}, 8..15 = K{base+16..+23},
//      base = (L<16) ? 0 : 8  (relative to k0).
//   B: lane L holds column (L&15) of B == row of W (since out = A*W^T); same K split.
//   C/D: lane L holds column (L&15); VGPR r <-> row r (lanes 0-15) / r+8 (lanes 16-31).
// ---------------------------------------------------------------------------
__global__ void __launch_bounds__(256)
gemm_wmma_bf16(const uint16_t* __restrict__ A, const uint16_t* __restrict__ W,
               const float* __restrict__ bias, float* __restrict__ outF,
               uint16_t* __restrict__ outB, int M, int outMode) {
    const int K = 256;
    const int gwave = blockIdx.x * (blockDim.x >> 5) + (threadIdx.x >> 5);
    const int numTiles = (M >> 4) * (H_CH >> 4);
    if (gwave >= numTiles) return;

    const int tileM = gwave >> 4;        // H_CH/16 = 16 column tiles
    const int tileN = gwave & 15;
    const int lane  = threadIdx.x & 31;
    const int mrow  = lane & 15;
    const int sel   = lane >> 4;         // 0: K base 0, 1: K base 8

    const size_t arow = (size_t)(tileM * 16 + mrow) * K;
    const size_t wrow = (size_t)(tileN * 16 + mrow) * K;   // B column = W row

    union Frag { uint4 u[2]; v16bf v; };

    v8f acc = {};
#pragma unroll
    for (int kk = 0; kk < 8; ++kk) {
        const int k0 = kk * 32;
        Frag fa, fb;
        fa.u[0] = *(const uint4*)(A + arow + k0 + sel * 8);
        fa.u[1] = *(const uint4*)(A + arow + k0 + 16 + sel * 8);
        fb.u[0] = *(const uint4*)(W + wrow + k0 + sel * 8);
        fb.u[1] = *(const uint4*)(W + wrow + k0 + 16 + sel * 8);
        acc = __builtin_amdgcn_wmma_f32_16x16x32_bf16(
            /*neg_a=*/false, fa.v, /*neg_b=*/false, fb.v,
            /*c_mod=*/(short)0, acc, /*reuse_a=*/false, /*reuse_b=*/false);
    }

    const int col = tileN * 16 + (lane & 15);
    const float bv = bias[col];
#pragma unroll
    for (int r = 0; r < 8; ++r) {
        const int row = tileM * 16 + ((lane < 16) ? r : r + 8);
        const float val = acc[r] + bv;
        if (outMode == 0) outF[(size_t)row * H_CH + col] = val;
        else              outB[(size_t)row * H_CH + col] = f2bf(val);
    }
}

// ---------------------------------------------------------------------------
// Sparse masked attention, one workgroup (256 threads) per node.
//   q fp32 [N,256], k/v bf16 [N,256], adj bitmask [N, 256 words]
//   out: attn bf16 [N,256]
// ---------------------------------------------------------------------------
__global__ void __launch_bounds__(256)
attn_sparse(const float* __restrict__ q, const uint16_t* __restrict__ kb,
            const uint16_t* __restrict__ vb, const uint32_t* __restrict__ adj,
            uint16_t* __restrict__ attnb) {
    __shared__ float qs[H_CH];
    __shared__ int   nbr[MAXD];
    __shared__ float sc[MAXD * HEADS];
    __shared__ float red[256];
    __shared__ float hinv[HEADS];
    __shared__ int   cnt;

    const int tid = threadIdx.x;
    const int n   = blockIdx.x;

    if (tid == 0) cnt = 0;
    __syncthreads();

    // extract neighbor list from this row's bitmask (dedup'd by construction)
    uint32_t w = adj[n * H_CH + tid];
    while (w) {
        int b = __ffs(w) - 1;
        w &= w - 1;
        int pos = atomicAdd(&cnt, 1);
        if (pos < MAXD) nbr[pos] = tid * 32 + b;
    }
    qs[tid] = q[(size_t)n * H_CH + tid];
    __syncthreads();
    const int deg = min(cnt, MAXD);

    // per-edge, per-head scores: dot(q[n,h,:], k[m,h,:]) / sqrt(64)
    const float scale = 0.125f;
    for (int idx = tid; idx < deg * HEADS; idx += 256) {
        const int j = idx >> 2, h = idx & 3;
        const int m = nbr[j];
        const uint4* kp = (const uint4*)(kb + (size_t)m * H_CH + h * HEAD_DIM);
        float s = 0.f;
#pragma unroll
        for (int c = 0; c < 8; ++c) {
            uint4 u = kp[c];
            const float* qp = &qs[h * HEAD_DIM + c * 8];
            s += qp[0] * bf2f(u.x & 0xffffu) + qp[1] * bf2f(u.x >> 16);
            s += qp[2] * bf2f(u.y & 0xffffu) + qp[3] * bf2f(u.y >> 16);
            s += qp[4] * bf2f(u.z & 0xffffu) + qp[5] * bf2f(u.z >> 16);
            s += qp[6] * bf2f(u.w & 0xffffu) + qp[7] * bf2f(u.w >> 16);
        }
        sc[idx] = s * scale;     // idx == j*4 + h
    }
    __syncthreads();

    // stable softmax per head over the neighbor list
    for (int h = 0; h < HEADS; ++h) {
        float mloc = -INFINITY;
        for (int j = tid; j < deg; j += 256) mloc = fmaxf(mloc, sc[j * 4 + h]);
        red[tid] = mloc; __syncthreads();
        for (int s = 128; s > 0; s >>= 1) {
            if (tid < s) red[tid] = fmaxf(red[tid], red[tid + s]);
            __syncthreads();
        }
        const float mx = red[0];
        __syncthreads();
        float sloc = 0.f;
        for (int j = tid; j < deg; j += 256) {
            float e = __expf(sc[j * 4 + h] - mx);
            sc[j * 4 + h] = e;
            sloc += e;
        }
        red[tid] = sloc; __syncthreads();
        for (int s = 128; s > 0; s >>= 1) {
            if (tid < s) red[tid] += red[tid + s];
            __syncthreads();
        }
        if (tid == 0) hinv[h] = (deg > 0) ? (1.0f / red[0]) : 0.0f;
        __syncthreads();
    }

    // aggregate V: thread tid <-> (head = tid/64, dim = tid%64); coalesced over tid per j
    const int h = tid >> 6;
    float acc = 0.f;
    for (int j = 0; j < deg; ++j)
        acc += sc[j * 4 + h] * bf2f((uint32_t)vb[(size_t)nbr[j] * H_CH + tid]);
    acc *= hinv[h];
    attnb[(size_t)n * H_CH + tid] = f2bf(acc);
}

// ---------------------------------------------------------------------------
// launch
// ---------------------------------------------------------------------------
extern "C" void kernel_launch(void* const* d_in, const int* in_sizes, int n_in,
                              void* d_out, int out_size, void* d_ws, size_t ws_size,
                              hipStream_t stream) {
    const float*     x    = (const float*)d_in[0];
    const long long* eidx = (const long long*)d_in[1];
    const float*     Wq   = (const float*)d_in[2];
    const float*     bq   = (const float*)d_in[3];
    const float*     Wk   = (const float*)d_in[4];
    const float*     bk   = (const float*)d_in[5];
    const float*     Wv   = (const float*)d_in[6];
    const float*     bv   = (const float*)d_in[7];
    const float*     Wo   = (const float*)d_in[8];
    const float*     bo   = (const float*)d_in[9];
    float*           out  = (float*)d_out;

    // workspace layout
    char* base = (char*)d_ws;
    const size_t MB = 1024 * 1024;
    uint16_t* xb    = (uint16_t*)(base + 0);          //  4 MiB  x bf16
    uint16_t* kb    = (uint16_t*)(base + 4 * MB);     //  4 MiB  K bf16
    uint16_t* vb    = (uint16_t*)(base + 8 * MB);     //  4 MiB  V bf16
    uint16_t* attnb = (uint16_t*)(base + 12 * MB);    //  4 MiB  attn out bf16
    float*    qf    = (float*)   (base + 16 * MB);    //  8 MiB  Q fp32
    uint32_t* adj   = (uint32_t*)(base + 24 * MB);    //  8 MiB  bitmask adjacency
    uint16_t* wqb   = (uint16_t*)(base + 32 * MB);
    uint16_t* wkb   = wqb + 65536;
    uint16_t* wvb   = wkb + 65536;
    uint16_t* wob   = wvb + 65536;

    const int NX = N_NODES * H_CH;   // 2097152
    const int NW = H_CH * H_CH;      // 65536

    // 1. zero adjacency bitmask (ws is not re-zeroed between calls)
    zero_u32<<<2048, 256, 0, stream>>>(adj, N_NODES * H_CH);

    // 2. f32 -> bf16 conversions
    cvt_f32_bf16<<<(NX + 255) / 256, 256, 0, stream>>>(x,  xb,  NX);
    cvt_f32_bf16<<<(NW + 255) / 256, 256, 0, stream>>>(Wq, wqb, NW);
    cvt_f32_bf16<<<(NW + 255) / 256, 256, 0, stream>>>(Wk, wkb, NW);
    cvt_f32_bf16<<<(NW + 255) / 256, 256, 0, stream>>>(Wv, wvb, NW);
    cvt_f32_bf16<<<(NW + 255) / 256, 256, 0, stream>>>(Wo, wob, NW);

    // 3. edges -> bitmask (dedupes duplicate edges, matching .at[].set(True))
    fill_adj<<<N_EDGES / 256, 256, 0, stream>>>(eidx, adj);

    // 4. projections via WMMA: 8192 tiles -> 1024 blocks x 8 waves
    gemm_wmma_bf16<<<1024, 256, 0, stream>>>(xb, wqb, bq, qf,      nullptr, N_NODES, 0);
    gemm_wmma_bf16<<<1024, 256, 0, stream>>>(xb, wkb, bk, nullptr, kb,      N_NODES, 1);
    gemm_wmma_bf16<<<1024, 256, 0, stream>>>(xb, wvb, bv, nullptr, vb,      N_NODES, 1);

    // 5. sparse masked softmax-attention, one workgroup per node
    attn_sparse<<<N_NODES, 256, 0, stream>>>(qf, kb, vb, adj, attnb);

    // 6. output projection via WMMA -> fp32 result
    gemm_wmma_bf16<<<1024, 256, 0, stream>>>(attnb, wob, bo, out, nullptr, N_NODES, 0);
}
